// LMSA_71347996721847
// MI455X (gfx1250) — matrix operand order repaired and checked by reading, not compiled, not run on hardware
//
#include <hip/hip_runtime.h>
#include <math.h>

// ---------------------------------------------------------------------------
// LMSA block for MI455X (gfx1250, wave32).
// Shapes: b=8, C=512, G=16, gc=32, h=w=80. N = b*G = 128 group-images.
// Strategy (bandwidth-bound, ~0.5 GB total traffic):
//   K1 rowmean   : gx.mean(axis=w)                      (1 pass over gx)
//   K2 xh        : rank-1 pw of rowmean -> x_h[n,c,y]
//   K3 sep chain : 4 rank-1 depthwise-separable stages per branch (tiny)
//   K4 pw_scalar : conv1 'u' contraction -> scalar field t[g,82,82]
//   K5 WMMA      : up conv as GEMM  D[32 x 16px] += A(up_w f16) x B(v1[c]*t)
//                  -> v_wmma_f32_16x16x32_f16, A staged in LDS
//   K6 s3        : u3-weighted dw3x3 conv scalar field   (1 pass over gx)
//   K7 stats1    : mean(x1_raw), mean(attn*x1_raw)       (1 pass over gx)
//   K8 stats2    : mean(s3), mean(attn*s3)
//   K9 softmix   : ECA + softmax collapse to coef2[n], g1[n,c] (rank-1 math)
//   K10 final    : weights = coef2*s3 + sum_c g1*x_h*gx; out = gx*sigmoid(w)
//                                                        (1 pass, 1 write)
// ---------------------------------------------------------------------------

typedef __attribute__((ext_vector_type(16))) _Float16 v16h;
typedef __attribute__((ext_vector_type(8)))  float    v8f;

__device__ __forceinline__ float wred32(float v) {
  #pragma unroll
  for (int o = 16; o > 0; o >>= 1) v += __shfl_down(v, o, 32);
  return v;
}

// ---- K1: rowmean[n,c,y] = mean_x gx[n,c,y,x]; one wave per row -------------
__global__ void rowmean_k(const float* __restrict__ x, float* __restrict__ rm) {
  int wid  = (blockIdx.x * blockDim.x + threadIdx.x) >> 5;   // (n*32+c)*80+y
  int lane = threadIdx.x & 31;
  if (wid >= 128 * 32 * 80) return;
  const float* row = x + (long)wid * 80;
  float v = 0.f;
  for (int j = lane; j < 80; j += 32) v += row[j];
  v = wred32(v);
  if (lane == 0) rm[wid] = v * (1.f / 80.f);
}

// ---- K2: x_h[n,c,y] = v[c] * sum_c' u[c'] * rowmean[n,c',y] ----------------
__global__ void xh_k(const float* __restrict__ rm, const float* __restrict__ u,
                     const float* __restrict__ v, float* __restrict__ xh) {
  int idx = blockIdx.x * blockDim.x + threadIdx.x;           // n*80 + y
  if (idx >= 128 * 80) return;
  int n = idx / 80, y = idx % 80;
  const float* r = rm + (long)n * 32 * 80 + y;
  float s = 0.f;
  #pragma unroll
  for (int c = 0; c < 32; ++c) s += u[c] * r[c * 80];
  float* o = xh + (long)n * 32 * 80 + y;
  #pragma unroll
  for (int c = 0; c < 32; ++c) o[c * 80] = v[c] * s;
}

// ---- K3: generic rank-1 depthwise-separable stage --------------------------
// out[n,c,p] = v[c] * sum_c' u[c'] * dwconv(dw[c'], in[n,c'])[p]
__global__ void sep_rank1(const float* __restrict__ in, long img_stride,
                          int Hi, int Wi, float* __restrict__ out, int Ho, int Wo,
                          const float* __restrict__ dw, int kh, int kw, int dil,
                          int padH, int padW,
                          const float* __restrict__ u, const float* __restrict__ v) {
  int n = blockIdx.y;
  int p = blockIdx.x * blockDim.x + threadIdx.x;
  if (p >= Ho * Wo) return;
  int y = p / Wo, x = p % Wo;
  const float* inn = in + (long)n * img_stride;
  float s = 0.f;
  for (int c = 0; c < 32; ++c) {
    const float* ic = inn + c * Hi * Wi;
    const float* wc = dw + c * kh * kw;
    float acc = 0.f;
    for (int i = 0; i < kh; ++i) {
      int yy = y - padH + i * dil;
      if (yy < 0 || yy >= Hi) continue;
      for (int j = 0; j < kw; ++j) {
        int xx = x - padW + j * dil;
        if (xx < 0 || xx >= Wi) continue;
        acc += ic[yy * Wi + xx] * wc[i * kw + j];
      }
    }
    s += u[c] * acc;
  }
  long ob = (long)n * 32 * Ho * Wo + p;
  for (int c = 0; c < 32; ++c) out[ob + (long)c * Ho * Wo] = v[c] * s;
}

// ---- K4: conv1 'u' contraction -> scalar field t[g=2n+branch, 82,82] -------
__global__ void pw_scalar(const float* __restrict__ in, const float* __restrict__ u1,
                          float* __restrict__ tfield, int branch) {
  int n = blockIdx.y;                                        // image 0..7
  int p = blockIdx.x * blockDim.x + threadIdx.x;
  if (p >= 82 * 82) return;
  const float* inn = in + (long)n * 32 * 6724 + p;
  float s = 0.f;
  #pragma unroll
  for (int c = 0; c < 32; ++c) s += u1[c] * inn[c * 6724];
  tfield[(long)(2 * n + branch) * 6724 + p] = s;
}

// ---- K5: up conv as WMMA GEMM ---------------------------------------------
// attn[g,o,y,x] = up_b[o] + sum_{ky,kx,c} up_w[o,c,ky,kx] * v1[c]*t[g,y+ky,x+kx]
// GEMM: M=32 (2 WMMA m-tiles), K = 9 taps * 32 ch, N = 16-pixel tiles.
// A (f16, LDS-staged) gfx1250 layout: elem e -> K=(e<8?e:e+8)+8*hi, M=lane&15
// B layout: elem e -> K=(e<8?e:e+8)+8*hi, N=lane&15 ; D: vgpr r -> M=r+8*hi, N=lane&15
__global__ __launch_bounds__(256)
void up_conv_wmma(const float* __restrict__ tfield, const float* __restrict__ up_w,
                  const float* __restrict__ v1, const float* __restrict__ up_b,
                  float* __restrict__ attn) {
  __shared__ _Float16 a_lds[9][32][32];   // [tap][o][c]  (18 KB of 320 KB LDS)
  __shared__ float v1s[32], bs[32];
  int tid = threadIdx.x;
  for (int i = tid; i < 9 * 32 * 32; i += 256) {
    int t = i >> 10, rem = i & 1023, o = rem >> 5, c = rem & 31;
    int ky = t / 3, kx = t % 3;
    a_lds[t][o][c] = (_Float16)up_w[((o * 32 + c) * 3 + ky) * 3 + kx];
  }
  if (tid < 32) { v1s[tid] = v1[tid]; bs[tid] = up_b[tid]; }
  __syncthreads();

  int g    = blockIdx.y;                  // 0..15 group image
  int wave = tid >> 5;
  int lane = tid & 31;
  int tile = blockIdx.x * 8 + wave;       // 0..399 : 80 rows x 5 tiles of 16 px
  int y  = tile / 5;
  int x0 = (tile % 5) * 16;
  int hi = lane >> 4;                     // lane half
  int ln = lane & 15;

  const float* tf = tfield + (long)g * 6724;

  float v1c[16];
  #pragma unroll
  for (int e = 0; e < 16; ++e) v1c[e] = v1s[((e < 8) ? e : (e + 8)) + hi * 8];

  v8f acc0 = {}; v8f acc1 = {};
  #pragma unroll
  for (int ky = 0; ky < 3; ++ky) {
    #pragma unroll
    for (int kx = 0; kx < 3; ++kx) {
      int tap = ky * 3 + kx;
      float tv = tf[(y + ky) * 82 + x0 + ln + kx];   // in-bounds by construction
      v16h bm;
      #pragma unroll
      for (int e = 0; e < 16; ++e) bm[e] = (_Float16)(v1c[e] * tv);
      v16h am0, am1;
      #pragma unroll
      for (int e = 0; e < 16; ++e) {
        int k = ((e < 8) ? e : (e + 8)) + hi * 8;
        am0[e] = a_lds[tap][ln][k];        // m-tile 0: out ch 0..15
        am1[e] = a_lds[tap][ln + 16][k];   // m-tile 1: out ch 16..31
      }
      acc0 = __builtin_amdgcn_wmma_f32_16x16x32_f16(false, am0, false, bm,
                                                    (short)0, acc0, false, false);
      acc1 = __builtin_amdgcn_wmma_f32_16x16x32_f16(false, am1, false, bm,
                                                    (short)0, acc1, false, false);
    }
  }
  int p = y * 80 + x0 + ln;
  float* outg = attn + (long)g * 32 * 6400;
  #pragma unroll
  for (int r = 0; r < 8; ++r) {
    int o0 = r + hi * 8;
    int o1 = o0 + 16;
    outg[(long)o0 * 6400 + p] = acc0[r] + bs[o0];
    outg[(long)o1 * 6400 + p] = acc1[r] + bs[o1];
  }
}

// ---- K6: s3[n,p] = sum_c u3[c] * dw3x3conv(gx[n,c])[p] ---------------------
__global__ void s3_k(const float* __restrict__ gx, const float* __restrict__ dw3,
                     const float* __restrict__ u3, float* __restrict__ s3) {
  long idx = (long)blockIdx.x * blockDim.x + threadIdx.x;    // n*6400+p
  if (idx >= 128L * 6400) return;
  int n = (int)(idx / 6400), p = (int)(idx % 6400);
  int y = p / 80, x = p % 80;
  const float* g = gx + (long)n * 32 * 6400;
  float s = 0.f;
  for (int c = 0; c < 32; ++c) {
    const float* ic = g + c * 6400;
    const float* wc = dw3 + c * 9;
    float acc = 0.f;
    #pragma unroll
    for (int i = 0; i < 3; ++i) {
      int yy = y - 1 + i;
      if (yy < 0 || yy >= 80) continue;
      #pragma unroll
      for (int j = 0; j < 3; ++j) {
        int xx = x - 1 + j;
        if (xx < 0 || xx >= 80) continue;
        acc += ic[yy * 80 + xx] * wc[i * 3 + j];
      }
    }
    s += u3[c] * acc;
  }
  s3[idx] = s;
}

// ---- K7: mean1[n,c]=mean_p(gx*x_h); A1[n,c]=mean_p(attn*gx*x_h) ------------
__global__ void stats1_k(const float* __restrict__ gx, const float* __restrict__ xh,
                         const float* __restrict__ attn,
                         float* __restrict__ mean1, float* __restrict__ A1) {
  int wid  = (blockIdx.x * blockDim.x + threadIdx.x) >> 5;   // n*32+c
  int lane = threadIdx.x & 31;
  if (wid >= 128 * 32) return;
  int n = wid >> 5, c = wid & 31, g = n & 15;
  const float* gc_ = gx + (long)wid * 6400;
  const float* xhc = xh + (long)wid * 80;
  const float* at  = attn + ((long)g * 32 + c) * 6400;
  float sv = 0.f, sa = 0.f;
  for (int p = lane; p < 6400; p += 32) {
    float val = gc_[p] * xhc[p / 80];
    sv += val;
    sa += at[p] * val;
  }
  sv = wred32(sv); sa = wred32(sa);
  if (lane == 0) { mean1[wid] = sv * (1.f / 6400.f); A1[wid] = sa * (1.f / 6400.f); }
}

// ---- K8a: A2[n,c] = mean_p attn[g,c,p]*s3[n,p] -----------------------------
__global__ void stats2_k(const float* __restrict__ s3, const float* __restrict__ attn,
                         float* __restrict__ A2) {
  int wid  = (blockIdx.x * blockDim.x + threadIdx.x) >> 5;
  int lane = threadIdx.x & 31;
  if (wid >= 128 * 32) return;
  int n = wid >> 5, c = wid & 31, g = n & 15;
  const float* sp = s3 + (long)n * 6400;
  const float* at = attn + ((long)g * 32 + c) * 6400;
  float sa = 0.f;
  for (int p = lane; p < 6400; p += 32) sa += at[p] * sp[p];
  sa = wred32(sa);
  if (lane == 0) A2[wid] = sa * (1.f / 6400.f);
}

// ---- K8b: ms3[n] = mean_p s3[n,p] ------------------------------------------
__global__ void ms3_k(const float* __restrict__ s3, float* __restrict__ ms3) {
  int wid  = (blockIdx.x * blockDim.x + threadIdx.x) >> 5;
  int lane = threadIdx.x & 31;
  if (wid >= 128) return;
  const float* sp = s3 + (long)wid * 6400;
  float v = 0.f;
  for (int p = lane; p < 6400; p += 32) v += sp[p];
  v = wred32(v);
  if (lane == 0) ms3[wid] = v * (1.f / 6400.f);
}

// ---- K9: ECA + softmax, collapsed via rank-1 algebra -----------------------
// x2[n,c,p] = f2[n,c]*s3[n,p] with f2 = v3*sig2;  x1[n,c,p] = sig1*gx*x_h
// weights[n,p] = coef2[n]*s3[n,p] + sum_c g1[n,c]*x_h[n,c,y]*gx[n,c,p]
__global__ void softmix_k(const float* __restrict__ mean1, const float* __restrict__ A1,
                          const float* __restrict__ ms3,  const float* __restrict__ A2,
                          const float* __restrict__ ecaw, const float* __restrict__ v3,
                          float* __restrict__ coef2, float* __restrict__ g1) {
  int n = blockIdx.x * blockDim.x + threadIdx.x;
  if (n >= 128) return;
  float w0 = ecaw[0], w1 = ecaw[1], w2 = ecaw[2];
  float yv[32], sig1[32], mv[32];
  #pragma unroll
  for (int c = 0; c < 32; ++c) yv[c] = mean1[n * 32 + c];
  #pragma unroll
  for (int c = 0; c < 32; ++c) {
    float l = c ? yv[c - 1] : 0.f, r = (c < 31) ? yv[c + 1] : 0.f;
    sig1[c] = 1.f / (1.f + expf(-(w0 * l + w1 * yv[c] + w2 * r)));
  }
  float mx = -1e30f;
  #pragma unroll
  for (int c = 0; c < 32; ++c) { mv[c] = sig1[c] * A1[n * 32 + c]; mx = fmaxf(mx, mv[c]); }
  float sum = 0.f;
  #pragma unroll
  for (int c = 0; c < 32; ++c) { mv[c] = expf(mv[c] - mx); sum += mv[c]; }
  float inv = 1.f / sum;                                     // x11 = mv*inv
  float ms = ms3[n];
  float f2[32];
  #pragma unroll
  for (int c = 0; c < 32; ++c) yv[c] = v3[c] * ms;           // y2
  #pragma unroll
  for (int c = 0; c < 32; ++c) {
    float l = c ? yv[c - 1] : 0.f, r = (c < 31) ? yv[c + 1] : 0.f;
    float s2 = 1.f / (1.f + expf(-(w0 * l + w1 * yv[c] + w2 * r)));
    f2[c] = v3[c] * s2;
  }
  float cf = 0.f;
  #pragma unroll
  for (int c = 0; c < 32; ++c) cf += (mv[c] * inv) * f2[c];
  coef2[n] = cf;
  float mv2[32]; float mx2 = -1e30f;
  #pragma unroll
  for (int c = 0; c < 32; ++c) { mv2[c] = f2[c] * A2[n * 32 + c]; mx2 = fmaxf(mx2, mv2[c]); }
  float sum2 = 0.f;
  #pragma unroll
  for (int c = 0; c < 32; ++c) { mv2[c] = expf(mv2[c] - mx2); sum2 += mv2[c]; }
  float inv2 = 1.f / sum2;                                   // x21 = mv2*inv2
  #pragma unroll
  for (int c = 0; c < 32; ++c) g1[n * 32 + c] = (mv2[c] * inv2) * sig1[c];
}

// ---- K10: fused weights + sigmoid gate + output ----------------------------
__global__ void final_k(const float* __restrict__ gx, const float* __restrict__ xh,
                        const float* __restrict__ s3, const float* __restrict__ coef2,
                        const float* __restrict__ g1, float* __restrict__ out) {
  long idx = (long)blockIdx.x * blockDim.x + threadIdx.x;    // n*6400+p
  if (idx >= 128L * 6400) return;
  int n = (int)(idx / 6400), p = (int)(idx % 6400), y = p / 80;
  const float* g   = gx + (long)n * 32 * 6400 + p;
  const float* xhn = xh + (long)n * 32 * 80 + y;
  const float* g1n = g1 + n * 32;
  float w = coef2[n] * s3[idx];
  float r[32];
  #pragma unroll
  for (int c = 0; c < 32; ++c) { r[c] = g[c * 6400]; w += g1n[c] * xhn[c * 80] * r[c]; }
  float sg = 1.f / (1.f + expf(-w));
  float* o = out + (long)n * 32 * 6400 + p;
  #pragma unroll
  for (int c = 0; c < 32; ++c) o[c * 6400] = r[c] * sg;
}

// ---------------------------------------------------------------------------
extern "C" void kernel_launch(void* const* d_in, const int* in_sizes, int n_in,
                              void* d_out, int out_size, void* d_ws, size_t ws_size,
                              hipStream_t stream) {
  (void)out_size; (void)ws_size;
  // Locate x (26,214,400 elements); params are the remaining 34 leaves in
  // jax-tree order (dict keys sorted recursively):
  //  0..23 branches[0..1].{c0h,c0v,sh,sv}.{dw,u,v}
  //  24 conv1.u 25 conv1.v 26 conv1x1.u 27 conv1x1.v
  //  28 conv3x3.dw 29 .u 30 .v   31 eca_w   32 up_b   33 up_w
  int x_idx = 0;
  for (int i = 0; i < n_in; ++i)
    if (in_sizes[i] == 8 * 512 * 6400) { x_idx = i; break; }
  const float* X = (const float*)d_in[x_idx];
  auto P = [&](int k) -> const float* {
    return (const float*)d_in[(x_idx == 0) ? (k + 1) : k];
  };

  float* ws      = (float*)d_ws;
  float* rowmean = ws;                      // 128*32*80
  float* xh      = rowmean + 327680;        // 128*32*80
  float* s3      = xh + 327680;             // 128*6400
  float* tf      = s3 + 819200;             // 16*82*82
  float* attn    = tf + 107584;             // 16*32*6400
  float* tmpA    = attn + 3276800;          // 8*32*82*82
  float* tmpB    = tmpA + 1721344;          // 8*32*82*82
  float* mean1   = tmpB + 1721344;          // 128*32
  float* A1      = mean1 + 4096;            // 128*32
  float* ms3     = A1 + 4096;               // 128
  float* A2      = ms3 + 128;               // 128*32
  float* coef2   = A2 + 4096;               // 128
  float* g1      = coef2 + 128;             // 128*32

  // K1/K2: width means + rank-1 conv1x1 -> x_h
  rowmean_k<<<40960, 256, 0, stream>>>(X, rowmean);
  xh_k<<<40, 256, 0, stream>>>(rowmean, P(26), P(27), xh);

  // K3/K4: attention branch chains (branch 0: groups 0,2..14; branch 1: odd)
  for (int s = 0; s < 2; ++s) {
    int base = s * 12;
    int ksz  = s ? 5 : 3;   // dilated sep kernel size
    int spad = s ? 5 : 3;
    sep_rank1<<<dim3(25, 8), 256, 0, stream>>>(X + (long)s * 204800, 409600L,
        80, 80, tmpA, 80, 80, P(base + 0), 1, 3, 1, 0, 1, P(base + 1), P(base + 2));
    sep_rank1<<<dim3(25, 8), 256, 0, stream>>>(tmpA, 204800L,
        80, 80, tmpB, 80, 80, P(base + 3), 3, 1, 1, 1, 0, P(base + 4), P(base + 5));
    sep_rank1<<<dim3(26, 8), 256, 0, stream>>>(tmpB, 204800L,
        80, 80, tmpA, 80, 82, P(base + 6), 1, ksz, 2, 0, spad, P(base + 7), P(base + 8));
    sep_rank1<<<dim3(27, 8), 256, 0, stream>>>(tmpA, 209920L,
        80, 82, tmpB, 82, 82, P(base + 9), ksz, 1, 2, spad, 0, P(base + 10), P(base + 11));
    pw_scalar<<<dim3(27, 8), 256, 0, stream>>>(tmpB, P(24), tf, s);
  }

  // K5: up conv via WMMA (16 images x 400 pixel-tiles; 8 waves/block)
  up_conv_wmma<<<dim3(50, 16), 256, 0, stream>>>(tf, P(33), P(25), P(32), attn);

  // K6..K8: scalar field + fused statistics
  s3_k<<<3200, 256, 0, stream>>>(X, P(28), P(29), s3);
  stats1_k<<<512, 256, 0, stream>>>(X, xh, attn, mean1, A1);
  stats2_k<<<512, 256, 0, stream>>>(s3, attn, A2);
  ms3_k<<<16, 256, 0, stream>>>(s3, ms3);

  // K9: ECA + softmax collapse
  softmix_k<<<1, 128, 0, stream>>>(mean1, A1, ms3, A2, P(31), P(30), coef2, g1);

  // K10: fused gate + output
  final_k<<<3200, 256, 0, stream>>>(X, xh, s3, coef2, g1, (float*)d_out);
}
